// ImplicitMultiHeadAttention_41979010351708
// MI455X (gfx1250) — matrix operand order repaired and checked by reading, not compiled, not run on hardware
//
#include <hip/hip_runtime.h>
#include <math.h>

// ---------------------------------------------------------------------------
// Problem constants (fixed by reference setup_inputs)
// ---------------------------------------------------------------------------
#define BATCH   4
#define SEQ     512
#define DMODEL  1024
#define NHEADS  16
#define HDIM    64          // DMODEL / NHEADS
#define MROWS   (BATCH*SEQ) // 2048
#define N3      (3*DMODEL)  // 3072
#define NITERS  10
#define SM_SCALE 0.125f     // (1024/16)^-0.5 = 1/8

typedef __bf16 bf16;
typedef __attribute__((ext_vector_type(16))) __bf16 bf16x16;
typedef __attribute__((ext_vector_type(8)))  __bf16 bf16x8;
typedef __attribute__((ext_vector_type(4)))  __bf16 bf16x4;
typedef __attribute__((ext_vector_type(8)))  float  f32x8;
typedef __attribute__((ext_vector_type(4)))  float  f32x4;

__device__ __forceinline__ bf16 f2bf(float f) {
    unsigned u = __float_as_uint(f);
    u += 0x7FFFu + ((u >> 16) & 1u);          // round-to-nearest-even
    unsigned short s = (unsigned short)(u >> 16);
    bf16 r; __builtin_memcpy(&r, &s, 2); return r;
}

// ---------------------------------------------------------------------------
// CDNA5 async global->LDS copy (ASYNCcnt-tracked), with sync fallback.
// Builtin signature (from probe diagnostics): param1 is
//   'int __attribute__((vector_size(16))) __device__ *'  (AS1 v4i32*)
// ---------------------------------------------------------------------------
#if __has_builtin(__builtin_amdgcn_global_load_async_to_lds_b128) && \
    __has_builtin(__builtin_amdgcn_s_wait_asynccnt)
#define USE_ASYNC_CP 1
#else
#define USE_ASYNC_CP 0
#endif

typedef int v4i32 __attribute__((__vector_size__(16)));
typedef __attribute__((address_space(1))) v4i32 as1_v4i32;
typedef __attribute__((address_space(3))) v4i32 as3_v4i32;

__device__ __forceinline__ void async_cp16(const bf16* g, bf16* l) {
#if USE_ASYNC_CP
    __builtin_amdgcn_global_load_async_to_lds_b128(
        (as1_v4i32*)(void*)g, (as3_v4i32*)(void*)l, 0, 0);
#else
    *(bf16x8*)l = *(const bf16x8*)g;
#endif
}
__device__ __forceinline__ void async_wait0() {
#if USE_ASYNC_CP
    __builtin_amdgcn_s_wait_asynccnt(0);
#endif
}

// --- CDNA5 bf16 WMMA fragment loads (wave32 lane layouts per ISA 7.12.2) ---
// A matrix 16x32: lane=(m,h); elems 0..7 = K[8h..8h+7], elems 8..15 = K[16+8h..]
__device__ __forceinline__ bf16x16 frag_a(const bf16* __restrict__ base, int ld, int lane) {
    int m = lane & 15, h = lane >> 4;
    const bf16* r = base + (size_t)m * ld;
    union { bf16x16 v; bf16x8 c[2]; } u;
    u.c[0] = *(const bf16x8*)(r + 8 * h);
    u.c[1] = *(const bf16x8*)(r + 16 + 8 * h);
    return u.v;
}
// B matrix 32x16 (stored as 16 N-rows of K elems): lane=(n,h); elems 0..7 = K[8h..8h+7], elems 8..15 = K[16+8h..]
__device__ __forceinline__ bf16x16 frag_b(const bf16* __restrict__ base, int ld, int lane) {
    int n = lane & 15, h = lane >> 4;
    const bf16* r = base + (size_t)n * ld;
    union { bf16x16 v; bf16x8 c[2]; } u;
    u.c[0] = *(const bf16x8*)(r + 8 * h);
    u.c[1] = *(const bf16x8*)(r + 16 + 8 * h);
    return u.v;
}

// ---------------------------------------------------------------------------
// Elementwise helpers
// ---------------------------------------------------------------------------
__global__ void cvt_f32_to_bf16(const float* __restrict__ in, bf16* __restrict__ out, int n) {
    int i = blockIdx.x * 256 + threadIdx.x;
    if (i < n) out[i] = f2bf(in[i]);
}
__global__ void zero_bf16(bf16* __restrict__ p, int n) {
    int i = blockIdx.x * 256 + threadIdx.x;
    if (i < n) { unsigned short z = 0; __builtin_memcpy(&p[i], &z, 2); }
}

// ---------------------------------------------------------------------------
// GEMM: C[M,N] = A[M,K](bf16) * W[N,K](bf16)^T  (+ bias f32)  -> f32 and/or bf16
// 128x128 block tile, 256 threads (8 waves), wave tile 32x64 (2x4 WMMA tiles).
// Double-buffered LDS stages of K=64, filled with async global->LDS copies.
// Dynamic LDS: 2 stages x (A(128x72) + B(128x72)) bf16 = 73728 bytes.
// ---------------------------------------------------------------------------
#define GEMM_LDT        72
#define GEMM_TILE_B     (128 * GEMM_LDT * 2)        // 18432 bytes per tile
#define GEMM_STAGE_B    (2 * GEMM_TILE_B)           // 36864 bytes per stage
#define GEMM_SMEM_BYTES (2 * GEMM_STAGE_B)          // 73728

template<bool ADD_BIAS, bool OUT_F32, bool OUT_BF16>
__global__ __launch_bounds__(256)
void gemm_bf16_wmma(const bf16* __restrict__ A, const bf16* __restrict__ W,
                    const float* __restrict__ bias,
                    float* __restrict__ Cf, bf16* __restrict__ Cb,
                    int M, int N, int K)
{
    extern __shared__ char gsm[];
    (void)M;
    const int tid  = threadIdx.x;
    const int lane = tid & 31;
    const int wv   = tid >> 5;
    const int wm   = wv & 3;              // wave row (4 along M)
    const int wn   = wv >> 2;             // wave col (2 along N)
    const int m0   = blockIdx.y * 128;
    const int n0   = blockIdx.x * 128;

    f32x8 acc[2][4] = {};

    auto stage_load = [&](int k0, int buf) {
        bf16* As = (bf16*)(gsm + buf * GEMM_STAGE_B);
        bf16* Bs = (bf16*)(gsm + buf * GEMM_STAGE_B + GEMM_TILE_B);
        #pragma unroll
        for (int i = 0; i < 4; ++i) {
            int idx = tid + i * 256;               // 0..1023
            int r = idx >> 3, c8 = (idx & 7) * 8;  // row, 8-elem chunk
            async_cp16(A + (size_t)(m0 + r) * K + k0 + c8, As + r * GEMM_LDT + c8);
            async_cp16(W + (size_t)(n0 + r) * K + k0 + c8, Bs + r * GEMM_LDT + c8);
        }
    };

    const int NS = K >> 6;                 // stages of K=64
    stage_load(0, 0);
    for (int ks = 0; ks < NS; ++ks) {
        async_wait0();                     // own stage-ks copies have landed
        __syncthreads();                   // all waves' copies visible; prev compute done
        if (ks + 1 < NS) stage_load((ks + 1) << 6, (ks + 1) & 1);

        const bf16* As = (const bf16*)(gsm + (ks & 1) * GEMM_STAGE_B);
        const bf16* Bs = (const bf16*)(gsm + (ks & 1) * GEMM_STAGE_B + GEMM_TILE_B);
        #pragma unroll
        for (int ksub = 0; ksub < 64; ksub += 32) {
            bf16x16 af[2], bfg[4];
            #pragma unroll
            for (int mt = 0; mt < 2; ++mt)
                af[mt] = frag_a(As + (wm * 32 + mt * 16) * GEMM_LDT + ksub, GEMM_LDT, lane);
            #pragma unroll
            for (int nt = 0; nt < 4; ++nt)
                bfg[nt] = frag_b(Bs + (wn * 64 + nt * 16) * GEMM_LDT + ksub, GEMM_LDT, lane);
            #pragma unroll
            for (int mt = 0; mt < 2; ++mt)
                #pragma unroll
                for (int nt = 0; nt < 4; ++nt)
                    acc[mt][nt] = __builtin_amdgcn_wmma_f32_16x16x32_bf16(
                        false, af[mt], false, bfg[nt], (short)0, acc[mt][nt], false, false);
        }
    }

    // epilogue: C layout -> lane holds column (lane&15), rows j + 8*(lane>>4)
    const int h = lane >> 4, nl = lane & 15;
    #pragma unroll
    for (int mt = 0; mt < 2; ++mt) {
        #pragma unroll
        for (int nt = 0; nt < 4; ++nt) {
            const int col = n0 + wn * 64 + nt * 16 + nl;
            #pragma unroll
            for (int j = 0; j < 8; ++j) {
                const int row = m0 + wm * 32 + mt * 16 + 8 * h + j;
                const size_t idx = (size_t)row * N + col;
                float v = acc[mt][nt][j];
                if (ADD_BIAS) v += bias[idx];
                if (OUT_F32)  Cf[idx] = v;
                if (OUT_BF16) Cb[idx] = f2bf(v);
            }
        }
    }
}

// ---------------------------------------------------------------------------
// Attention: per block = one (b, head, 64-query tile); 128 threads (4 waves),
// each wave owns 16 query rows. Full 512-wide score rows in LDS (f32),
// parallel vectorized softmax, normalized P (bf16) @ V via WMMA against
// LDS-transposed V.
// Dynamic LDS layout (bytes):
//   Vt  [64][520] bf16 : 0      .. 66560
//   Sc  [64][516] f32  : 66560  .. 198656
//   Pb  [64][520] bf16 : 198656 .. 265216
//   rowmax[64] f32     : 265216
//   rowinv[64] f32     : 265472
//   pmax[128]  f32     : 265728
//   psum[128]  f32     : 266240 .. 266752
// ---------------------------------------------------------------------------
#define ATT_SMEM_BYTES 266752

__global__ __launch_bounds__(128)
void attn_kernel(const bf16* __restrict__ qkv,           // [MROWS, 3*DMODEL]
                 const unsigned char* __restrict__ mask, // [BATCH, SEQ]
                 bf16* __restrict__ outb)                // [MROWS, DMODEL]
{
    extern __shared__ char smem[];
    bf16*  Vt     = (bf16*)(smem);                       // [64][520]
    float* Sc     = (float*)(smem + 66560);              // [64][516]
    bf16*  Pb     = (bf16*)(smem + 198656);              // [64][520]
    float* rowmax = (float*)(smem + 265216);
    float* rowinv = (float*)(smem + 265472);
    float* pmax   = (float*)(smem + 265728);
    float* psum   = (float*)(smem + 266240);

    const int tid  = threadIdx.x;
    const int lane = tid & 31;
    const int wv   = tid >> 5;                // 0..3, wave = 16 query rows
    const int bx   = blockIdx.x;
    const int qt   = bx & 7;                  // query tile (64 rows each)
    const int head = (bx >> 3) & 15;
    const int b    = bx >> 7;

    const int    q0   = qt * 64;              // first query row of block
    const size_t rowQ = (size_t)(b * SEQ);    // row base in qkv

    // ---- phase 1: load V[512,64] transposed into Vt[64][520] --------------
    for (int i = 0; i < 32; ++i) {
        int idx = tid + i * 128;              // 0..4095, 8 elems each
        int n = idx >> 3, d8 = (idx & 7) * 8;
        bf16x8 vvv = *(const bf16x8*)(qkv + (rowQ + n) * N3 + 2 * DMODEL + head * HDIM + d8);
        #pragma unroll
        for (int j = 0; j < 8; ++j) Vt[(d8 + j) * 520 + n] = vvv[j];
    }

    // ---- phase 2: scores S = (Q K^T)*scale + mask -> Sc -------------------
    const bf16* qbase = qkv + (rowQ + q0 + wv * 16) * N3 + head * HDIM;
    bf16x16 aq0 = frag_a(qbase,      N3, lane);   // hd K = 0..31
    bf16x16 aq1 = frag_a(qbase + 32, N3, lane);   // hd K = 32..63
    const int h  = lane >> 4, nl = lane & 15;

    for (int n0 = 0; n0 < SEQ; n0 += 16) {
        const bf16* kbase = qkv + (rowQ + n0) * N3 + DMODEL + head * HDIM;
        bf16x16 bk0 = frag_b(kbase,      N3, lane);
        bf16x16 bk1 = frag_b(kbase + 32, N3, lane);
        f32x8 sc = {};
        sc = __builtin_amdgcn_wmma_f32_16x16x32_bf16(false, aq0, false, bk0, (short)0, sc, false, false);
        sc = __builtin_amdgcn_wmma_f32_16x16x32_bf16(false, aq1, false, bk1, (short)0, sc, false, false);
        const int   n    = n0 + nl;
        const float madd = mask[(size_t)b * SEQ + n] ? 0.0f : -1e9f;
        #pragma unroll
        for (int j = 0; j < 8; ++j) {
            int m = wv * 16 + 8 * h + j;
            Sc[m * 516 + n] = sc[j] * SM_SCALE + madd;
        }
    }
    __syncthreads();

    // ---- phase 3: softmax stats (2 threads per row, f32x4 vector scans) ---
    const int srow = tid >> 1, shalf = tid & 1;
    const f32x4* sv = (const f32x4*)(Sc + srow * 516 + shalf * 256);
    {
        f32x4 v = sv[0];
        float mx = fmaxf(fmaxf(v[0], v[1]), fmaxf(v[2], v[3]));
        for (int i = 1; i < 64; ++i) {
            v = sv[i];
            mx = fmaxf(mx, fmaxf(fmaxf(v[0], v[1]), fmaxf(v[2], v[3])));
        }
        pmax[tid] = mx;
    }
    __syncthreads();
    const float mrow = fmaxf(pmax[srow * 2], pmax[srow * 2 + 1]);
    {
        float s = 0.0f;
        for (int i = 0; i < 64; ++i) {
            f32x4 v = sv[i];
            s += __expf(v[0] - mrow) + __expf(v[1] - mrow) +
                 __expf(v[2] - mrow) + __expf(v[3] - mrow);
        }
        psum[tid] = s;
    }
    __syncthreads();
    if (shalf == 0) {
        rowmax[srow] = mrow;
        rowinv[srow] = 1.0f / (psum[srow * 2] + psum[srow * 2 + 1]);
    }
    __syncthreads();

    // normalized probabilities in bf16 (vectorized: 4 elems / store)
    for (int i = 0; i < 64; ++i) {
        int idx = tid + i * 128;              // 0..8191 vec4 units
        int m = idx >> 7, c4 = (idx & 127) * 4;
        f32x4 v = *(const f32x4*)(Sc + m * 516 + c4);
        float mr = rowmax[m], ir = rowinv[m];
        bf16x4 p;
        p[0] = f2bf(__expf(v[0] - mr) * ir);
        p[1] = f2bf(__expf(v[1] - mr) * ir);
        p[2] = f2bf(__expf(v[2] - mr) * ir);
        p[3] = f2bf(__expf(v[3] - mr) * ir);
        *(bf16x4*)(Pb + m * 520 + c4) = p;
    }
    __syncthreads();

    // ---- phase 4: O = P @ V ----------------------------------------------
    f32x8 oacc[4] = {};
    for (int n0 = 0; n0 < SEQ; n0 += 32) {
        bf16x16 ap = frag_a(Pb + wv * 16 * 520 + n0, 520, lane);
        #pragma unroll
        for (int dt = 0; dt < 4; ++dt) {
            bf16x16 bvv = frag_b(Vt + dt * 16 * 520 + n0, 520, lane);
            oacc[dt] = __builtin_amdgcn_wmma_f32_16x16x32_bf16(
                false, ap, false, bvv, (short)0, oacc[dt], false, false);
        }
    }

    // ---- phase 5: store O (bf16) -----------------------------------------
    #pragma unroll
    for (int dt = 0; dt < 4; ++dt) {
        const int col = head * HDIM + dt * 16 + nl;
        #pragma unroll
        for (int j = 0; j < 8; ++j) {
            const int row = q0 + wv * 16 + 8 * h + j;
            outb[(rowQ + row) * DMODEL + col] = f2bf(oacc[dt][j]);
        }
    }
}

// ---------------------------------------------------------------------------
// Host-side orchestration
// ---------------------------------------------------------------------------
extern "C" void kernel_launch(void* const* d_in, const int* in_sizes, int n_in,
                              void* d_out, int out_size, void* d_ws, size_t ws_size,
                              hipStream_t stream) {
    (void)in_sizes; (void)n_in; (void)out_size; (void)ws_size;
    const float* A_w   = (const float*)d_in[0];          // [3072,1024]
    const float* Wqkv  = (const float*)d_in[1];          // [3072,1024]
    const float* Wo    = (const float*)d_in[2];          // [1024,1024]
    const float* x     = (const float*)d_in[3];          // [4,512,1024]
    const unsigned char* mask = (const unsigned char*)d_in[4]; // [4,512] bool
    float* zout = (float*)d_out;                         // [4,512,1024]

    char* ws = (char*)d_ws;
    size_t off = 0;
    auto alloc = [&](size_t bytes) { size_t r = off; off += (bytes + 255) & ~(size_t)255; return r; };
    bf16* x_bf    = (bf16*)(ws + alloc((size_t)MROWS * DMODEL * 2));
    bf16* Wqkv_bf = (bf16*)(ws + alloc((size_t)N3 * DMODEL * 2));
    bf16* A_bf    = (bf16*)(ws + alloc((size_t)N3 * DMODEL * 2));
    bf16* Wo_bf   = (bf16*)(ws + alloc((size_t)DMODEL * DMODEL * 2));
    float* U      = (float*)(ws + alloc((size_t)MROWS * N3 * 4));
    bf16* qkv_bf  = (bf16*)(ws + alloc((size_t)MROWS * N3 * 2));
    bf16* out_bf  = (bf16*)(ws + alloc((size_t)MROWS * DMODEL * 2));
    bf16* z_bf    = (bf16*)(ws + alloc((size_t)MROWS * DMODEL * 2));

    const int nXD = MROWS * DMODEL;      // 2097152
    const int nW3 = N3 * DMODEL;         // 3145728
    const int nWo = DMODEL * DMODEL;     // 1048576

    cvt_f32_to_bf16<<<(nXD + 255) / 256, 256, 0, stream>>>(x,    x_bf,    nXD);
    cvt_f32_to_bf16<<<(nW3 + 255) / 256, 256, 0, stream>>>(Wqkv, Wqkv_bf, nW3);
    cvt_f32_to_bf16<<<(nW3 + 255) / 256, 256, 0, stream>>>(A_w,  A_bf,    nW3);
    cvt_f32_to_bf16<<<(nWo + 255) / 256, 256, 0, stream>>>(Wo,   Wo_bf,   nWo);
    zero_bf16      <<<(nXD + 255) / 256, 256, 0, stream>>>(z_bf, nXD);

    // one-time: U = x @ Wqkv^T  (iteration-invariant half of the qkv GEMM)
    gemm_bf16_wmma<false, true, false>
        <<<dim3(N3 / 128, MROWS / 128), 256, GEMM_SMEM_BYTES, stream>>>(
        x_bf, Wqkv_bf, nullptr, U, nullptr, MROWS, N3, DMODEL);

    for (int it = 0; it < NITERS; ++it) {
        // qkv = bf16(U + z @ A_w^T)
        gemm_bf16_wmma<true, false, true>
            <<<dim3(N3 / 128, MROWS / 128), 256, GEMM_SMEM_BYTES, stream>>>(
            z_bf, A_bf, U, nullptr, qkv_bf, MROWS, N3, DMODEL);
        // multi-head attention -> out_bf
        attn_kernel<<<BATCH * NHEADS * (SEQ / 64), 128, ATT_SMEM_BYTES, stream>>>(
            qkv_bf, mask, out_bf);
        // z = out @ Wo^T  (f32 to d_out, bf16 for next iteration)
        gemm_bf16_wmma<false, true, true>
            <<<dim3(DMODEL / 128, MROWS / 128), 256, GEMM_SMEM_BYTES, stream>>>(
            out_bf, Wo_bf, nullptr, zout, z_bf, MROWS, DMODEL, DMODEL);
    }
}
